// SpGAT_modified_59992103190881
// MI455X (gfx1250) — compile-verified, matrix operand order checked by reading, not run on hardware
//
#include <hip/hip_runtime.h>
#include <hip/hip_bf16.h>
#include <math.h>

typedef __attribute__((ext_vector_type(16))) _Float16 v16h;
typedef __attribute__((ext_vector_type(8)))  _Float16 v8h;
typedef __attribute__((ext_vector_type(8)))  float    v8f;

__device__ __forceinline__ float lrelu02(float x){ return x > 0.f ? x : 0.2f*x; }
__device__ __forceinline__ float elu1(float x){ return x > 0.f ? x : (expf(x)-1.f); }

// ---------------- utility kernels ----------------

__global__ void k_zero(float* p, long n){
  long i = (long)blockIdx.x*blockDim.x + threadIdx.x;
  long stride = (long)gridDim.x*blockDim.x;
  for (; i < n; i += stride) p[i] = 0.f;
}

// Pack f32 matrix a (128 x K) into WMMA A-fragment order (f16).
// Fragment (mt,kt) = 16x32 tile at (M=16*mt, K=32*kt); per-lane v16h layout per ISA:
// lanes 0-15:  elems 0..7 = K+0..7,  8..15 = K+16..23  (M = lane)
// lanes 16-31: elems 0..7 = K+8..15, 8..15 = K+24..31  (M = lane-16)
__global__ void k_pack_a(const float* __restrict__ a, _Float16* __restrict__ ap, int K, int NKT){
  int tid = blockIdx.x*blockDim.x + threadIdx.x;
  int nfrag = 8*NKT;
  if (tid >= nfrag*32) return;
  int lane = tid & 31, frag = tid >> 5;
  int kt = frag % NKT, mt = frag / NKT;
  int m  = 16*mt + (lane & 15);
  int kb = 32*kt + ((lane & 16) ? 8 : 0);
  _Float16* dst = ap + (size_t)tid*16;
  #pragma unroll
  for (int i = 0; i < 8; i++){
    dst[i]     = (_Float16)a[(size_t)m*K + kb + i];
    dst[8 + i] = (_Float16)a[(size_t)m*K + kb + 16 + i];
  }
}

// c = a2 (1x128) @ a (128xK)  -> K vector
__global__ void k_avec(const float* __restrict__ a2, const float* __restrict__ a,
                       float* __restrict__ c, int K){
  int j = blockIdx.x*blockDim.x + threadIdx.x;
  if (j >= K) return;
  float s = 0.f;
  for (int i = 0; i < 128; i++) s += a2[i]*a[(size_t)i*K + j];
  c[j] = s;
}

__global__ void k_gemm(const float* __restrict__ A, const float* __restrict__ B,
                       float* __restrict__ C, int M, int K, int N){
  int t = blockIdx.x*blockDim.x + threadIdx.x;
  if (t >= M*N) return;
  int i = t / N, j = t % N;
  float s = 0.f;
  for (int k = 0; k < K; k++) s += A[(size_t)i*K + k]*B[(size_t)k*N + j];
  C[(size_t)i*N + j] = s;
}

__global__ void k_gemm_gather(const float* __restrict__ X, const int* __restrict__ rows,
                              const float* __restrict__ W, float* __restrict__ Y,
                              int n, int K, int N){
  int t = blockIdx.x*blockDim.x + threadIdx.x;
  if (t >= n*N) return;
  int i = t / N, j = t % N;
  const float* x = X + (size_t)rows[i]*K;
  float s = 0.f;
  for (int k = 0; k < K; k++) s += x[k]*W[(size_t)k*N + j];
  Y[(size_t)i*N + j] = s;
}

__global__ void k_norm_mid(float* __restrict__ x, const float* __restrict__ rs0,
                           const float* __restrict__ rs1, int n){
  int t = blockIdx.x*blockDim.x + threadIdx.x;
  if (t >= n*256) return;
  int i = t >> 8, c = t & 255;
  float r = (c < 128 ? rs0[i] : rs1[i]) + 1e-12f;
  x[t] = elu1(x[t]/r);
}

__global__ void k_norm_last(float* __restrict__ h, const float* __restrict__ rs, int n){
  int t = blockIdx.x*blockDim.x + threadIdx.x;
  if (t >= n*128) return;
  int i = t >> 7;
  h[t] = elu1(h[t]/(rs[i] + 1e-12f));
}

__global__ void k_set_rows(const float* __restrict__ src, const int* __restrict__ rows,
                           float* __restrict__ dst, int n, int w){
  int t = blockIdx.x*blockDim.x + threadIdx.x;
  if (t >= n*w) return;
  int i = t / w, j = t % w;
  dst[(size_t)rows[i]*w + j] = src[t];
}

__global__ void k_scatter_out(const float* __restrict__ hfin, const int* __restrict__ target_cont,
                              const float* __restrict__ tgtemb, const int* __restrict__ scatter_target,
                              float* __restrict__ out, int ntgt){
  int t = blockIdx.x*blockDim.x + threadIdx.x;
  if (t >= ntgt*128) return;
  int i = t >> 7, j = t & 127;
  out[(size_t)scatter_target[i]*128 + j] = hfin[(size_t)target_cont[i]*128 + j] + tgtemb[t];
}

// ---------------- WMMA edge-GAT kernel ----------------

template<int KF>
__device__ __forceinline__ void stage_rows32(_Float16 (*sh)[KF], int lane,
    const float* base, const int* idx, int w, int off, int gc, int nedges){
  for (int n = 0; n < 32; n++){
    int e = gc*32 + n; if (e >= nedges) e = nedges - 1;
    const float* row = base + (size_t)idx[e]*w;
    for (int tt = lane*4; tt < w; tt += 128){
      float4 v = *(const float4*)(row + tt);
      sh[n][off+tt+0] = (_Float16)v.x;
      sh[n][off+tt+1] = (_Float16)v.y;
      sh[n][off+tt+2] = (_Float16)v.z;
      sh[n][off+tt+3] = (_Float16)v.w;
    }
  }
}

// Single-wave workgroup. Per trip: 32 edges -> stage f16 features in LDS, then
// edge_m (128 x 32) = a (128 x KF) @ h (KF x 32) via WMMA with 2 N-tiles sharing
// every A-fragment load (2x arithmetic intensity vs 16-edge tiles). Attention
// coef e = exp(-leakyrelu((a2@a) . h)); e*edge_m scattered via f32 atomics.
template<int KF, int NKT, int NL>
__global__ __launch_bounds__(32) void k_edge_gat(
    const float* __restrict__ baseA, const int* __restrict__ idxA, int wA,
    const float* __restrict__ baseB, const int* __restrict__ idxB, int wB,
    const float* __restrict__ baseC, const int* __restrict__ idxC, int wC,
    const int* __restrict__ tgt_cont,
    const _Float16* __restrict__ ap0, const float* __restrict__ c0,
    float* __restrict__ out0, int ostr0, int ocol0, float* __restrict__ rs0,
    const _Float16* __restrict__ ap1, const float* __restrict__ c1,
    float* __restrict__ out1, int ostr1, int ocol1, float* __restrict__ rs1,
    int nedges)
{
  __shared__ __align__(16) _Float16 sh[32][KF];
  __shared__ float sE[2][32];
  const int lane = threadIdx.x;
  const int wave = blockIdx.x;
  const int nwaves = gridDim.x;
  const int ngroups = (nedges + 31) >> 5;
  const int trips = (ngroups + nwaves - 1)/nwaves;

  for (int t = 0; t < trips; t++){
    int g = wave + t*nwaves;
    bool active = g < ngroups;
    int gc = active ? g : 0;

    // stage features (f32 -> f16) into LDS, B-matrix staging area
    stage_rows32<KF>(sh, lane, baseA, idxA, wA, 0,       gc, nedges);
    stage_rows32<KF>(sh, lane, baseB, idxB, wB, wA,      gc, nedges);
    stage_rows32<KF>(sh, lane, baseC, idxC, wC, wA + wB, gc, nedges);
    __syncthreads();

    // attention logits: s = (a2 @ a) . h  (linearity of a2 @ (a @ h)); lane = edge
    {
      int e = gc*32 + lane;
      bool eact = active && (e < nedges);
      float s0 = 0.f, s1 = 0.f;
      for (int k = 0; k < KF; k++){
        float v = (float)sh[lane][k];
        s0 += c0[k]*v;
        if (NL == 2) s1 += c1[k]*v;
      }
      float e0 = eact ? expf(-lrelu02(s0)) : 0.f;
      sE[0][lane] = e0;
      float e1 = 0.f;
      if (NL == 2){ e1 = eact ? expf(-lrelu02(s1)) : 0.f; sE[1][lane] = e1; }
      if (eact){
        int tg = tgt_cont[e];
        atomicAdd(rs0 + tg, e0);
        if (NL == 2) atomicAdd(rs1 + tg, e1);
      }
    }
    __syncthreads();

    // WMMA: 8 M-tiles x 2 N-tiles, A fragments shared across both N-tiles
    {
      const int ncol = lane & 15;
      const int kA   = (lane & 16) ? 8 : 0;
      const int mrow = (lane & 16) ? 8 : 0;
      const int ei0 = gc*32 + ncol;
      const int ei1 = gc*32 + 16 + ncol;
      const bool ea0 = active && (ei0 < nedges);
      const bool ea1 = active && (ei1 < nedges);
      const int tg0 = ea0 ? tgt_cont[ei0] : 0;
      const int tg1 = ea1 ? tgt_cont[ei1] : 0;
      for (int L = 0; L < NL; ++L){
        const _Float16* ap = (L == 0) ? ap0 : ap1;
        float* out = (L == 0) ? out0 : out1;
        const int ostr = (L == 0) ? ostr0 : ostr1;
        const int ocol = (L == 0) ? ocol0 : ocol1;
        const float w0 = sE[L][ncol];
        const float w1 = sE[L][16 + ncol];
        v8f acc0[8] = {};
        v8f acc1[8] = {};
        for (int kt = 0; kt < NKT; kt++){
          v16h bf0, bf1;
          {
            const _Float16* pp = &sh[ncol][32*kt + kA];
            v8h lo = *(const v8h*)pp;
            v8h hi = *(const v8h*)(pp + 16);
            #pragma unroll
            for (int i = 0; i < 8; i++){ bf0[i] = lo[i]; bf0[8+i] = hi[i]; }
          }
          {
            const _Float16* pp = &sh[16 + ncol][32*kt + kA];
            v8h lo = *(const v8h*)pp;
            v8h hi = *(const v8h*)(pp + 16);
            #pragma unroll
            for (int i = 0; i < 8; i++){ bf1[i] = lo[i]; bf1[8+i] = hi[i]; }
          }
          #pragma unroll
          for (int mt = 0; mt < 8; mt++){
            v16h af = *(const v16h*)(ap + ((size_t)(mt*NKT + kt)*32 + lane)*16);
            acc0[mt] = __builtin_amdgcn_wmma_f32_16x16x32_f16(
                false, af, false, bf0, (short)0, acc0[mt], false, false);
            acc1[mt] = __builtin_amdgcn_wmma_f32_16x16x32_f16(
                false, af, false, bf1, (short)0, acc1[mt], false, false);
          }
        }
        if (ea0){
          #pragma unroll
          for (int mt = 0; mt < 8; mt++){
            #pragma unroll
            for (int r = 0; r < 8; r++){
              int m = 16*mt + r + mrow;
              atomicAdd(out + (size_t)tg0*ostr + ocol + m, acc0[mt][r]*w0);
            }
          }
        }
        if (ea1){
          #pragma unroll
          for (int mt = 0; mt < 8; mt++){
            #pragma unroll
            for (int r = 0; r < 8; r++){
              int m = 16*mt + r + mrow;
              atomicAdd(out + (size_t)tg1*ostr + ocol + m, acc1[mt][r]*w1);
            }
          }
        }
      }
    }
    __syncthreads();
  }
}

// ---------------- launcher ----------------

extern "C" void kernel_launch(void* const* d_in, const int* in_sizes, int n_in,
                              void* d_out, int out_size, void* d_ws, size_t ws_size,
                              hipStream_t stream) {
  const float* ent   = (const float*)d_in[0];   // 50000 x 128
  const float* rel   = (const float*)d_in[1];   // 500 x 128
  const float* W1    = (const float*)d_in[2];   // 128 x 128
  const float* W3    = (const float*)d_in[3];   // 128 x 128
  const float* Wsrc  = (const float*)d_in[4];   // 128 x 256
  const float* Wtgt  = (const float*)d_in[5];   // 128 x 128
  const float* a_h0  = (const float*)d_in[6];   // 128 x 384
  const float* a2_h0 = (const float*)d_in[7];   // 1 x 128
  const float* a_h1  = (const float*)d_in[8];
  const float* a2_h1 = (const float*)d_in[9];
  const float* a_l   = (const float*)d_in[10];  // 128 x 640
  const float* a2_l  = (const float*)d_in[11];
  const int* etype   = (const int*)d_in[12];
  const int* etgt    = (const int*)d_in[13];
  const int* esrc    = (const int*)d_in[14];
  const int* tgtc    = (const int*)d_in[15];
  const int* srcc    = (const int*)d_in[16];
  const int* scat_t  = (const int*)d_in[17];
  const int* targ_c  = (const int*)d_in[18];
  const int* scat_s  = (const int*)d_in[19];
  const int* src_c   = (const int*)d_in[20];

  const int E      = in_sizes[12];
  const int NT     = in_sizes[17];
  const int NS     = in_sizes[19];
  const int nsc    = NT + NS;          // n_scatter = |unique(tgt)| + |setdiff(src,tgt)|
  const int NREL   = in_sizes[1] / 128;
  const int NNODES = in_sizes[0] / 128;

  // workspace layout (256B aligned slots)
  char* p = (char*)d_ws;
  auto alloc = [&](size_t bytes)->char*{
    char* r = p; p += (bytes + 255) & ~(size_t)255; return r;
  };
  float*    c0     = (float*)   alloc(384*4);
  float*    c1     = (float*)   alloc(384*4);
  float*    cl     = (float*)   alloc(640*4);
  _Float16* a0p    = (_Float16*)alloc((size_t)8*12*512*2);
  _Float16* a1p    = (_Float16*)alloc((size_t)8*12*512*2);
  _Float16* alp    = (_Float16*)alloc((size_t)8*20*512*2);
  float*    outrel = (float*)   alloc((size_t)NREL*128*4);
  float*    srcemb = (float*)   alloc((size_t)NS*256*4);
  float*    tgtemb = (float*)   alloc((size_t)NT*128*4);
  float*    xmid   = (float*)   alloc((size_t)nsc*256*4);   // accumulators start here
  float*    hlast  = (float*)   alloc((size_t)nsc*128*4);
  float*    rs0    = (float*)   alloc((size_t)nsc*4);
  float*    rs1    = (float*)   alloc((size_t)nsc*4);
  float*    rsl    = (float*)   alloc((size_t)nsc*4);
  long nzero = (long)((p - (char*)xmid)/4);

  k_zero<<<1024, 256, 0, stream>>>(xmid, nzero);

  k_pack_a<<<(8*12*32 + 127)/128, 128, 0, stream>>>(a_h0, a0p, 384, 12);
  k_pack_a<<<(8*12*32 + 127)/128, 128, 0, stream>>>(a_h1, a1p, 384, 12);
  k_pack_a<<<(8*20*32 + 127)/128, 128, 0, stream>>>(a_l,  alp, 640, 20);
  k_avec<<<(384 + 127)/128, 128, 0, stream>>>(a2_h0, a_h0, c0, 384);
  k_avec<<<(384 + 127)/128, 128, 0, stream>>>(a2_h1, a_h1, c1, 384);
  k_avec<<<(640 + 127)/128, 128, 0, stream>>>(a2_l,  a_l,  cl, 640);

  // out_rel = rel @ W1 ; second output = out_rel @ W3
  k_gemm<<<(NREL*128 + 127)/128, 128, 0, stream>>>(rel, W1, outrel, NREL, 128, 128);
  k_gemm<<<(NREL*128 + 127)/128, 128, 0, stream>>>(outrel, W3,
        (float*)d_out + (size_t)NNODES*128, NREL, 128, 128);

  if (NS > 0)
    k_gemm_gather<<<((size_t)NS*256 + 127)/128, 128, 0, stream>>>(ent, scat_s, Wsrc, srcemb, NS, 128, 256);
  if (NT > 0)
    k_gemm_gather<<<((size_t)NT*128 + 127)/128, 128, 0, stream>>>(ent, scat_t, Wtgt, tgtemb, NT, 128, 128);

  // layer 1: two heads over edge features [x[tgt] | x[src] | rel[type]] (K=384)
  k_edge_gat<384,12,2><<<2048, 32, 0, stream>>>(
      ent, etgt, 128,  ent, esrc, 128,  rel, etype, 128,  tgtc,
      a0p, c0, xmid, 256, 0,   rs0,
      a1p, c1, xmid, 256, 128, rs1,
      E);
  k_norm_mid<<<((size_t)nsc*256 + 255)/256, 256, 0, stream>>>(xmid, rs0, rs1, nsc);
  if (NS > 0)
    k_set_rows<<<((size_t)NS*256 + 255)/256, 256, 0, stream>>>(srcemb, src_c, xmid, NS, 256);

  // layer 2: [xmid[tgt_cont] | xmid[src_cont] | out_rel[type]] (K=640)
  k_edge_gat<640,20,1><<<2048, 32, 0, stream>>>(
      xmid, tgtc, 256,  xmid, srcc, 256,  outrel, etype, 128,  tgtc,
      alp, cl, hlast, 128, 0, rsl,
      alp, cl, hlast, 128, 0, rsl,
      E);
  k_norm_last<<<((size_t)nsc*128 + 255)/256, 256, 0, stream>>>(hlast, rsl, nsc);

  // ent_new = entity_embeddings; rows[scatter_target] = hlast[target_cont] + tgt_emb
  hipMemcpyAsync(d_out, (const void*)ent, (size_t)NNODES*128*4, hipMemcpyDeviceToDevice, stream);
  if (NT > 0)
    k_scatter_out<<<((size_t)NT*128 + 255)/256, 256, 0, stream>>>(hlast, targ_c, tgtemb, scat_t,
        (float*)d_out, NT);
}
